// MessagePassingLayer_31653908972328
// MI455X (gfx1250) — matrix operand order repaired and checked by reading, not compiled, not run on hardware
//
#include <hip/hip_runtime.h>

#define DD 128

typedef __attribute__((ext_vector_type(16))) _Float16 v16h;
typedef __attribute__((ext_vector_type(8)))  _Float16 v8h;
typedef __attribute__((ext_vector_type(8)))  float    v8f;
typedef __attribute__((ext_vector_type(4)))  float    v4f;
typedef __attribute__((ext_vector_type(4)))  unsigned int v4u;
typedef __attribute__((ext_vector_type(8)))  int      v8i;
typedef __attribute__((ext_vector_type(4)))  int      v4i;

#if defined(__has_builtin)
#if __has_builtin(__builtin_amdgcn_tensor_load_to_lds) && __has_builtin(__builtin_amdgcn_s_wait_tensorcnt)
#define HAVE_TDM 1
#endif
#endif
#ifndef HAVE_TDM
#define HAVE_TDM 0
#endif

__device__ __forceinline__ float lrelu(float x) { return x > 0.0f ? x : 0.01f * x; }

__device__ __forceinline__ v8f wmma_f16(v16h a, v16h b, v8f c) {
  // D = A(16x32 f16) * B(32x16 f16) + C(16x16 f32)
  return __builtin_amdgcn_wmma_f32_16x16x32_f16(false, a, false, b, (short)0, c, false, false);
}

// A fragment (16x32 f16) from fp32 memory. Contiguous k order: lane = h*16+m holds
// K = h*16 + [0..7] in elements 0..7 and K = h*16 + 8 + [0..7] in elements 8..15.
// p0 -> 8 consecutive fp32 at first run, p1 -> 8 consecutive fp32 at second run.
__device__ __forceinline__ v16h a_frag_f32(const float* p0, const float* p1) {
  v4f x0 = *(const v4f*)(p0);
  v4f x1 = *(const v4f*)(p0 + 4);
  v4f y0 = *(const v4f*)(p1);
  v4f y1 = *(const v4f*)(p1 + 4);
  v16h a;
#pragma unroll
  for (int i = 0; i < 4; ++i) {
    a[i]      = (_Float16)x0[i];
    a[4 + i]  = (_Float16)x1[i];
    a[8 + i]  = (_Float16)y0[i];
    a[12 + i] = (_Float16)y1[i];
  }
  return a;
}

// Same A layout but from f16 memory (two runs of 8 halves).
__device__ __forceinline__ v16h a_frag_h(const _Float16* p0, const _Float16* p1) {
  v8h lo = *(const v8h*)p0;
  v8h hi = *(const v8h*)p1;
  v16h a;
#pragma unroll
  for (int i = 0; i < 8; ++i) { a[i] = lo[i]; a[8 + i] = hi[i]; }
  return a;
}

// B fragment (32x16 f16): lane = h*16+n holds K = h*16 + [0..15] of column n.
// With W stored transposed ([n][k], f16), that's 16 contiguous halves.
__device__ __forceinline__ v16h b_frag_h(const _Float16* p) {
  return *(const v16h*)p;
}

#if HAVE_TDM
// 2D tensor-DMA load into LDS. Elements are 2 bytes. Tile: tile_rows lines of
// tile_k elements, line stride = stride_elems; packed contiguously in LDS.
__device__ __forceinline__ void tdm_load_2d(void* lds_dst, const void* gsrc,
                                            unsigned tile_k, unsigned tile_rows,
                                            unsigned tensor_k, unsigned tensor_rows,
                                            unsigned stride_elems)
{
  unsigned long long ga = (unsigned long long)gsrc;
  unsigned lds_off = (unsigned)(unsigned long long)lds_dst;
  v4u g0;
  g0.x = 1u;                                            // count=1, no gather
  g0.y = lds_off;                                       // lds_addr
  g0.z = (unsigned)ga;                                  // global_addr[31:0]
  g0.w = (unsigned)((ga >> 32) & 0x01ffffffull) | 0x80000000u; // addr msbs | type=2
  v8i g1;
  g1[0] = 0x00010000;                                   // data_size = 2B
  g1[1] = (int)((tensor_k & 0xffffu) << 16);            // tensor_dim0[15:0]
  g1[2] = (int)(((tensor_k >> 16) & 0xffffu) | ((tensor_rows & 0xffffu) << 16));
  g1[3] = (int)(((tensor_rows >> 16) & 0xffffu) | ((tile_k & 0xffffu) << 16));
  g1[4] = (int)(tile_rows & 0xffffu);                   // tile_dim1 (tile_dim2=0)
  g1[5] = (int)stride_elems;                            // tensor_dim0_stride[31:0]
  g1[6] = 0;
  g1[7] = 0;
  v4i g2 = {0, 0, 0, 0};                                // dims 3/4 unused (2D)
  v4i g3 = {0, 0, 0, 0};
  v8i g4 = {0, 0, 0, 0, 0, 0, 0, 0};                    // extra group (clang-23 form)
  __builtin_amdgcn_tensor_load_to_lds(g0, g1, g2, g3, g4, 0);
}
#endif

// ---------------- prep kernels ----------------

// dst[n*K + k] = src[k*128 + n]  (fp32 [K,128] -> f16 transposed [128,K])
__global__ void cvt_w_kernel(const float* __restrict__ src, _Float16* __restrict__ dst, int K) {
  int i = blockIdx.x * blockDim.x + threadIdx.x;
  if (i < K * DD) {
    int n = i / K, k = i - n * K;
    dst[i] = (_Float16)src[(size_t)k * DD + n];
  }
}

__global__ void zero_kernel(float* __restrict__ p, long n) {
  long i = (long)blockIdx.x * blockDim.x + threadIdx.x;
  if (i < n) p[i] = 0.0f;
}

// per-graph column sums (segments are contiguous, uniform counts)
__global__ __launch_bounds__(128) void graph_sum_kernel(
    const float* __restrict__ nodes, const float* __restrict__ edges,
    float* __restrict__ summed_nodes, float* __restrict__ summed_edges,
    int n_per_g, int e_per_g)
{
  int g = blockIdx.x, t = threadIdx.x;
  const float* p = nodes + (size_t)g * n_per_g * DD + t;
  float s = 0.0f;
  for (int i = 0; i < n_per_g; ++i) s += p[(size_t)i * DD];
  summed_nodes[(size_t)g * DD + t] = s;
  const float* q = edges + (size_t)g * e_per_g * DD + t;
  float s2 = 0.0f;
  for (int i = 0; i < e_per_g; ++i) s2 += q[(size_t)i * DD];
  summed_edges[(size_t)g * DD + t] = s2;
}

// ---------------- edge kernel (the big one) ----------------
// 256 threads = 8 waves, 16 edges per wave, 128 edges per workgroup.
// W0 K-chunks are double-buffered in LDS via TDM so the DMA for chunk kk+1
// overlaps the 16 WMMAs of chunk kk (TENSORcnt completes in order -> wait<=2).
__global__ __launch_bounds__(256) void edge_kernel(
    const float* __restrict__ nodes, const float* __restrict__ edges,
    const int* __restrict__ senders, const int* __restrict__ receivers,
    const _Float16* __restrict__ nodeW0T, const _Float16* __restrict__ edgeW0T,
    const _Float16* __restrict__ nodeW1T, const _Float16* __restrict__ edgeW1T,
    const float* __restrict__ node_b0, const float* __restrict__ node_b1,
    const float* __restrict__ edge_b0, const float* __restrict__ edge_b1,
    float* __restrict__ new_nodes, float* __restrict__ new_edges)
{
  __shared__ __align__(64) _Float16 sW0[2][2][128 * 32]; // [buf][mlp] W0 K-chunk (32KB)
  __shared__ __align__(64) _Float16 sY[8][16 * 128];     // per-wave y0 tile (32KB)

  const int tid  = threadIdx.x;
  const int wave = tid >> 5;
  const int lane = tid & 31;
  const int h    = lane >> 4;     // half of wave
  const int ln   = lane & 15;     // row m for A / col n for B,C
  const int ebase = blockIdx.x * 128 + wave * 16;
  const int e = ebase + ln;

  const float* rowS = nodes + (size_t)senders[e]   * DD;
  const float* rowR = nodes + (size_t)receivers[e] * DD;
  const float* rowE = edges + (size_t)e * DD;

  // C init = bias (same value for all 8 rows held by a lane; col = nn*16+ln)
  v8f accN[8], accE[8];
#pragma unroll
  for (int nn = 0; nn < 8; ++nn) {
    float bn = node_b0[nn * 16 + ln];
    float be = edge_b0[nn * 16 + ln];
#pragma unroll
    for (int r = 0; r < 8; ++r) { accN[nn][r] = bn; accE[nn][r] = be; }
  }

#if HAVE_TDM
  // prefetch chunk 0
  if (wave == 0) {
    tdm_load_2d(&sW0[0][0][0], nodeW0T, 32, 128, 384, 128, 384);
    tdm_load_2d(&sW0[0][1][0], edgeW0T, 32, 128, 384, 128, 384);
  }
#endif

  // -------- layer 0: concat[E,384] x W0[384,128], K in 12 chunks of 32 --------
  for (int kk = 0; kk < 12; ++kk) {
    const int buf = kk & 1;
    // all waves done reading buffer (kk+1)&1 from iteration kk-1
    __syncthreads();
#if HAVE_TDM
    if (wave == 0) {
      if (kk + 1 < 12) {
        // prefetch next chunk into the other buffer, then wait for current:
        // 2 ops just issued are outstanding; <=2 means chunk kk's 2 ops retired.
        tdm_load_2d(&sW0[buf ^ 1][0][0], nodeW0T + (kk + 1) * 32, 32, 128, 384, 128, 384);
        tdm_load_2d(&sW0[buf ^ 1][1][0], edgeW0T + (kk + 1) * 32, 32, 128, 384, 128, 384);
        __builtin_amdgcn_s_wait_tensorcnt(2);
      } else {
        __builtin_amdgcn_s_wait_tensorcnt(0);
      }
    }
#else
    for (int i = tid; i < 2 * 128; i += 256) {
      const _Float16* s = (i < 128 ? nodeW0T : edgeW0T) + (size_t)(i & 127) * 384 + kk * 32;
      _Float16* dptr = &sW0[buf][i >> 7][(i & 127) * 32];
#pragma unroll
      for (int q = 0; q < 4; ++q) ((v8h*)dptr)[q] = ((const v8h*)s)[q];
    }
#endif
    __syncthreads();  // chunk kk data visible in sW0[buf]

    const int k0 = kk * 32 + h * 16;
    const int k1 = kk * 32 + h * 16 + 8;
    const float* p0 = (k0 < 128) ? rowS + k0 : (k0 < 256) ? rowR + (k0 - 128) : rowE + (k0 - 256);
    const float* p1 = (k1 < 128) ? rowS + k1 : (k1 < 256) ? rowR + (k1 - 128) : rowE + (k1 - 256);
    v16h a = a_frag_f32(p0, p1);
#pragma unroll
    for (int nn = 0; nn < 8; ++nn) {
      accN[nn] = wmma_f16(a, b_frag_h(&sW0[buf][0][(nn * 16 + ln) * 32 + h * 16]), accN[nn]);
      accE[nn] = wmma_f16(a, b_frag_h(&sW0[buf][1][(nn * 16 + ln) * 32 + h * 16]), accE[nn]);
    }
  }

  _Float16* sYw = &sY[wave][0];

  // -------- node MLP layer 1 -> atomic scatter-add into new_nodes --------
#pragma unroll
  for (int nn = 0; nn < 8; ++nn)
#pragma unroll
    for (int r = 0; r < 8; ++r)
      sYw[(r + 8 * h) * 128 + nn * 16 + ln] = (_Float16)lrelu(accN[nn][r]);

  {
    v8f acc2[8];
#pragma unroll
    for (int nn = 0; nn < 8; ++nn) {
      float bv = node_b1[nn * 16 + ln];
#pragma unroll
      for (int r = 0; r < 8; ++r) acc2[nn][r] = bv;
    }
#pragma unroll
    for (int cc = 0; cc < 4; ++cc) {
      v16h a1 = a_frag_h(sYw + ln * 128 + cc * 32 + h * 16,
                         sYw + ln * 128 + cc * 32 + h * 16 + 8);
#pragma unroll
      for (int nn = 0; nn < 8; ++nn)
        acc2[nn] = wmma_f16(a1, b_frag_h(nodeW1T + (size_t)(nn * 16 + ln) * DD + cc * 32 + h * 16), acc2[nn]);
    }
#pragma unroll
    for (int nn = 0; nn < 8; ++nn)
#pragma unroll
      for (int r = 0; r < 8; ++r) {
        int mm = r + 8 * h;
        int col = nn * 16 + ln;
        atomicAdd(&new_nodes[(size_t)receivers[ebase + mm] * DD + col], lrelu(acc2[nn][r]));
      }
  }

  // -------- edge MLP layer 1 -> new_edges --------
#pragma unroll
  for (int nn = 0; nn < 8; ++nn)
#pragma unroll
    for (int r = 0; r < 8; ++r)
      sYw[(r + 8 * h) * 128 + nn * 16 + ln] = (_Float16)lrelu(accE[nn][r]);

  {
    v8f acc2[8];
#pragma unroll
    for (int nn = 0; nn < 8; ++nn) {
      float bv = edge_b1[nn * 16 + ln];
#pragma unroll
      for (int r = 0; r < 8; ++r) acc2[nn][r] = bv;
    }
#pragma unroll
    for (int cc = 0; cc < 4; ++cc) {
      v16h a1 = a_frag_h(sYw + ln * 128 + cc * 32 + h * 16,
                         sYw + ln * 128 + cc * 32 + h * 16 + 8);
#pragma unroll
      for (int nn = 0; nn < 8; ++nn)
        acc2[nn] = wmma_f16(a1, b_frag_h(edgeW1T + (size_t)(nn * 16 + ln) * DD + cc * 32 + h * 16), acc2[nn]);
    }
#pragma unroll
    for (int nn = 0; nn < 8; ++nn)
#pragma unroll
      for (int r = 0; r < 8; ++r) {
        int mm = r + 8 * h;
        int col = nn * 16 + ln;
        new_edges[(size_t)(ebase + mm) * DD + col] = lrelu(acc2[nn][r]);
      }
  }
}

// ---------------- globals kernel ----------------
// 3 sub-MLPs (glob / gnode / gedge), then final MLP over the 384-wide concat.
// Final layer-0 is accumulated per 128-wide section so no concat buffer needed.
__global__ __launch_bounds__(256) void global_mlp_kernel(
    const float* __restrict__ globals_,
    const float* __restrict__ summed_nodes,
    const float* __restrict__ summed_edges,
    const _Float16* __restrict__ globW0T,  const _Float16* __restrict__ globW1T,
    const _Float16* __restrict__ gnodeW0T, const _Float16* __restrict__ gnodeW1T,
    const _Float16* __restrict__ gedgeW0T, const _Float16* __restrict__ gedgeW1T,
    const _Float16* __restrict__ finW0T,   const _Float16* __restrict__ finW1T,
    const float* __restrict__ glob_b0,  const float* __restrict__ glob_b1,
    const float* __restrict__ gnode_b0, const float* __restrict__ gnode_b1,
    const float* __restrict__ gedge_b0, const float* __restrict__ gedge_b1,
    const float* __restrict__ fin_b0,   const float* __restrict__ fin_b1,
    float* __restrict__ new_global)
{
  __shared__ __align__(64) _Float16 sY[8][16 * 128];
  const int tid  = threadIdx.x;
  const int wave = tid >> 5;
  const int lane = tid & 31;
  const int h    = lane >> 4;
  const int ln   = lane & 15;
  const int rbase = (blockIdx.x * 8 + wave) * 16;
  _Float16* sYw = &sY[wave][0];

  const float* srcs[3]     = {globals_, summed_nodes, summed_edges};
  const _Float16* W0s[3]   = {globW0T, gnodeW0T, gedgeW0T};
  const _Float16* W1s[3]   = {globW1T, gnodeW1T, gedgeW1T};
  const float* b0s[3]      = {glob_b0, gnode_b0, gedge_b0};
  const float* b1s[3]      = {glob_b1, gnode_b1, gedge_b1};

  v8f accF[8];
#pragma unroll
  for (int nn = 0; nn < 8; ++nn) {
    float bv = fin_b0[nn * 16 + ln];
#pragma unroll
    for (int r = 0; r < 8; ++r) accF[nn][r] = bv;
  }

#pragma unroll
  for (int j = 0; j < 3; ++j) {
    const float* row = srcs[j] + (size_t)(rbase + ln) * DD;
    v8f acc[8];
#pragma unroll
    for (int nn = 0; nn < 8; ++nn) {
      float bv = b0s[j][nn * 16 + ln];
#pragma unroll
      for (int r = 0; r < 8; ++r) acc[nn][r] = bv;
    }
#pragma unroll
    for (int cc = 0; cc < 4; ++cc) {
      v16h a = a_frag_f32(row + cc * 32 + h * 16, row + cc * 32 + h * 16 + 8);
#pragma unroll
      for (int nn = 0; nn < 8; ++nn)
        acc[nn] = wmma_f16(a, b_frag_h(W0s[j] + (size_t)(nn * 16 + ln) * DD + cc * 32 + h * 16), acc[nn]);
    }
    // y0 -> LDS
#pragma unroll
    for (int nn = 0; nn < 8; ++nn)
#pragma unroll
      for (int r = 0; r < 8; ++r)
        sYw[(r + 8 * h) * 128 + nn * 16 + ln] = (_Float16)lrelu(acc[nn][r]);
    // layer 1
    v8f acc2[8];
#pragma unroll
    for (int nn = 0; nn < 8; ++nn) {
      float bv = b1s[j][nn * 16 + ln];
#pragma unroll
      for (int r = 0; r < 8; ++r) acc2[nn][r] = bv;
    }
#pragma unroll
    for (int cc = 0; cc < 4; ++cc) {
      v16h a1 = a_frag_h(sYw + ln * 128 + cc * 32 + h * 16,
                         sYw + ln * 128 + cc * 32 + h * 16 + 8);
#pragma unroll
      for (int nn = 0; nn < 8; ++nn)
        acc2[nn] = wmma_f16(a1, b_frag_h(W1s[j] + (size_t)(nn * 16 + ln) * DD + cc * 32 + h * 16), acc2[nn]);
    }
    // sub-MLP output -> LDS (overwrite y0), then accumulate final layer-0 section j
#pragma unroll
    for (int nn = 0; nn < 8; ++nn)
#pragma unroll
      for (int r = 0; r < 8; ++r)
        sYw[(r + 8 * h) * 128 + nn * 16 + ln] = (_Float16)lrelu(acc2[nn][r]);
#pragma unroll
    for (int cc = 0; cc < 4; ++cc) {
      v16h a2 = a_frag_h(sYw + ln * 128 + cc * 32 + h * 16,
                         sYw + ln * 128 + cc * 32 + h * 16 + 8);
#pragma unroll
      for (int nn = 0; nn < 8; ++nn)
        accF[nn] = wmma_f16(a2, b_frag_h(finW0T + (size_t)(nn * 16 + ln) * 384 + j * 128 + cc * 32 + h * 16), accF[nn]);
    }
  }

  // final layer 1
#pragma unroll
  for (int nn = 0; nn < 8; ++nn)
#pragma unroll
    for (int r = 0; r < 8; ++r)
      sYw[(r + 8 * h) * 128 + nn * 16 + ln] = (_Float16)lrelu(accF[nn][r]);
  v8f acc2[8];
#pragma unroll
  for (int nn = 0; nn < 8; ++nn) {
    float bv = fin_b1[nn * 16 + ln];
#pragma unroll
    for (int r = 0; r < 8; ++r) acc2[nn][r] = bv;
  }
#pragma unroll
  for (int cc = 0; cc < 4; ++cc) {
    v16h a1 = a_frag_h(sYw + ln * 128 + cc * 32 + h * 16,
                       sYw + ln * 128 + cc * 32 + h * 16 + 8);
#pragma unroll
    for (int nn = 0; nn < 8; ++nn)
      acc2[nn] = wmma_f16(a1, b_frag_h(finW1T + (size_t)(nn * 16 + ln) * DD + cc * 32 + h * 16), acc2[nn]);
  }
#pragma unroll
  for (int nn = 0; nn < 8; ++nn)
#pragma unroll
    for (int r = 0; r < 8; ++r)
      new_global[(size_t)(rbase + r + 8 * h) * DD + nn * 16 + ln] = lrelu(acc2[nn][r]);
}

// ---------------- host entry ----------------
extern "C" void kernel_launch(void* const* d_in, const int* in_sizes, int n_in,
                              void* d_out, int out_size, void* d_ws, size_t ws_size,
                              hipStream_t stream)
{
  (void)n_in; (void)out_size; (void)ws_size;
  const float* nodes     = (const float*)d_in[0];
  const float* edges     = (const float*)d_in[1];
  const float* globals_  = (const float*)d_in[2];
  const int*   senders   = (const int*)d_in[3];
  const int*   receivers = (const int*)d_in[4];
  const float* node_W0 = (const float*)d_in[7];
  const float* node_b0 = (const float*)d_in[8];
  const float* node_W1 = (const float*)d_in[9];
  const float* node_b1 = (const float*)d_in[10];
  const float* edge_W0 = (const float*)d_in[11];
  const float* edge_b0 = (const float*)d_in[12];
  const float* edge_W1 = (const float*)d_in[13];
  const float* edge_b1 = (const float*)d_in[14];
  const float* glob_W0 = (const float*)d_in[15];
  const float* glob_b0 = (const float*)d_in[16];
  const float* glob_W1 = (const float*)d_in[17];
  const float* glob_b1 = (const float*)d_in[18];
  const float* gnode_W0 = (const float*)d_in[19];
  const float* gnode_b0 = (const float*)d_in[20];
  const float* gnode_W1 = (const float*)d_in[21];
  const float* gnode_b1 = (const float*)d_in[22];
  const float* gedge_W0 = (const float*)d_in[23];
  const float* gedge_b0 = (const float*)d_in[24];
  const float* gedge_W1 = (const float*)d_in[25];
  const float* gedge_b1 = (const float*)d_in[26];
  const float* fin_W0 = (const float*)d_in[27];
  const float* fin_b0 = (const float*)d_in[28];
  const float* fin_W1 = (const float*)d_in[29];
  const float* fin_b1 = (const float*)d_in[30];

  const int N = in_sizes[0] / DD;   // 51200
  const int E = in_sizes[1] / DD;   // 409600
  const int G = in_sizes[2] / DD;   // 256

  // workspace layout (halves, then floats)
  _Float16* wsh = (_Float16*)d_ws;
  _Float16* nodeW0T = wsh + 0;        // 384*128
  _Float16* edgeW0T = wsh + 49152;    // 384*128
  _Float16* nodeW1T = wsh + 98304;    // 128*128
  _Float16* edgeW1T = wsh + 114688;
  _Float16* globW0T = wsh + 131072;
  _Float16* globW1T = wsh + 147456;
  _Float16* gnodeW0T = wsh + 163840;
  _Float16* gnodeW1T = wsh + 180224;
  _Float16* gedgeW0T = wsh + 196608;
  _Float16* gedgeW1T = wsh + 212992;
  _Float16* finW0T  = wsh + 229376;   // 384*128
  _Float16* finW1T  = wsh + 278528;   // 128*128  -> end 294912 halves
  float* summed_nodes = (float*)((char*)d_ws + (size_t)294912 * 2);
  float* summed_edges = summed_nodes + (size_t)G * DD;

  float* out_nodes  = (float*)d_out;
  float* out_edges  = out_nodes + (size_t)N * DD;
  float* out_global = out_edges + (size_t)E * DD;

  // weight transpose+convert
  cvt_w_kernel<<<(384 * DD + 255) / 256, 256, 0, stream>>>(node_W0, nodeW0T, 384);
  cvt_w_kernel<<<(384 * DD + 255) / 256, 256, 0, stream>>>(edge_W0, edgeW0T, 384);
  cvt_w_kernel<<<(128 * DD + 255) / 256, 256, 0, stream>>>(node_W1, nodeW1T, 128);
  cvt_w_kernel<<<(128 * DD + 255) / 256, 256, 0, stream>>>(edge_W1, edgeW1T, 128);
  cvt_w_kernel<<<(128 * DD + 255) / 256, 256, 0, stream>>>(glob_W0, globW0T, 128);
  cvt_w_kernel<<<(128 * DD + 255) / 256, 256, 0, stream>>>(glob_W1, globW1T, 128);
  cvt_w_kernel<<<(128 * DD + 255) / 256, 256, 0, stream>>>(gnode_W0, gnodeW0T, 128);
  cvt_w_kernel<<<(128 * DD + 255) / 256, 256, 0, stream>>>(gnode_W1, gnodeW1T, 128);
  cvt_w_kernel<<<(128 * DD + 255) / 256, 256, 0, stream>>>(gedge_W0, gedgeW0T, 128);
  cvt_w_kernel<<<(128 * DD + 255) / 256, 256, 0, stream>>>(gedge_W1, gedgeW1T, 128);
  cvt_w_kernel<<<(384 * DD + 255) / 256, 256, 0, stream>>>(fin_W0, finW0T, 384);
  cvt_w_kernel<<<(128 * DD + 255) / 256, 256, 0, stream>>>(fin_W1, finW1T, 128);

  // zero the atomic-accumulated output region
  long nzero = (long)N * DD;
  zero_kernel<<<(int)((nzero + 255) / 256), 256, 0, stream>>>(out_nodes, nzero);

  graph_sum_kernel<<<G, 128, 0, stream>>>(nodes, edges, summed_nodes, summed_edges,
                                          N / G, E / G);

  edge_kernel<<<E / 128, 256, 0, stream>>>(nodes, edges, senders, receivers,
                                           nodeW0T, edgeW0T, nodeW1T, edgeW1T,
                                           node_b0, node_b1, edge_b0, edge_b1,
                                           out_nodes, out_edges);

  global_mlp_kernel<<<G / 128, 256, 0, stream>>>(globals_, summed_nodes, summed_edges,
                                                 globW0T, globW1T, gnodeW0T, gnodeW1T,
                                                 gedgeW0T, gedgeW1T, finW0T, finW1T,
                                                 glob_b0, glob_b1, gnode_b0, gnode_b1,
                                                 gedge_b0, gedge_b1, fin_b0, fin_b1,
                                                 out_global);
}
